// CWSA_Layer_58007828300558
// MI455X (gfx1250) — matrix operand order repaired and checked by reading, not compiled, not run on hardware
//
#include <hip/hip_runtime.h>

// Problem constants (match reference)
#define BB 4
#define CC 256
#define C4 64
#define NN 4096
// fold 1/sqrt(FACTOR) = 1/sqrt(8) into Q once, so E = Q^T Q / 8 out of WMMA.
#define QSCALE 0.35355339059327373f
// constant softmax shift used during accumulation (E diag ~ 8, off-diag std ~1;
// exp(e - 8) cannot overflow; exact normalizers reconstructed at the end).
#define ESHIFT 8.0f

typedef __attribute__((ext_vector_type(16))) __bf16 v16bf;
typedef __attribute__((ext_vector_type(8)))  __bf16 v8bf;
typedef __attribute__((ext_vector_type(8)))  float  v8f;
typedef __attribute__((ext_vector_type(4)))  unsigned int v4u;
typedef __attribute__((ext_vector_type(8)))  int v8i;
typedef __attribute__((ext_vector_type(4)))  int v4i;

union V16U { v16bf v; v8bf h[2]; };

#if __has_builtin(__builtin_amdgcn_tensor_load_to_lds)
#define HAVE_TDM 1
#endif

static __device__ __forceinline__ v8f wmma_bf16(v16bf a, v16bf b, v8f c) {
  return __builtin_amdgcn_wmma_f32_16x16x32_bf16(false, a, false, b, (short)0, c, false, false);
}

// A operand (16x32 bf16, M=rows, K=c): lane holds row (lane&15); lanes 0-15
// carry K = kh*32+0..15, lanes 16-31 carry kh*32+16..31.
static __device__ __forceinline__ v16bf load_a_frag(const __bf16* row, int kh, int half) {
  V16U r;
  r.h[0] = *(const v8bf*)(row + kh * 32 + half * 16);
  r.h[1] = *(const v8bf*)(row + kh * 32 + half * 16 + 8);
  return r.v;
}

// B operand (32x16 bf16, K=c, N=cols): lane holds column (lane&15); lanes 0-15
// carry K = kh*32+0..15, lanes 16-31 carry kh*32+16..31 -> one 32B chunk.
static __device__ __forceinline__ v16bf load_b_frag(const __bf16* row, int kh, int half) {
  V16U r;
  const v8bf* p = (const v8bf*)(row + kh * 32 + half * 16);
  r.h[0] = p[0];
  r.h[1] = p[1];
  return r.v;
}

// TDM: post a 2-D D# moving a 32-row x 64-col bf16 tile (4 KB, rows contiguous
// with stride 64) from global Qt into LDS. Tracked by TENSORcnt.
static __device__ __forceinline__ void tdm_load_qtile(const __bf16* gsrc,
                                                      unsigned lds_byte_addr) {
#ifdef HAVE_TDM
  unsigned long long ga = (unsigned long long)(uintptr_t)gsrc;
  v4u g0;
  g0[0] = 1u;                       // count=1 (valid user descriptor)
  g0[1] = lds_byte_addr;            // LDS destination (byte address)
  g0[2] = (unsigned)ga;             // global_addr[31:0]
  g0[3] = (unsigned)((ga >> 32) & 0x01FFFFFFu) | 0x80000000u; // addr[56:32]|type=2
  v8i g1;
  g1[0] = 0x00010000;               // workgroup_mask=0, data_size=1 (2 bytes)
  g1[1] = (int)(64u << 16);         // tensor_dim0 = 64 (bits 79:48)
  g1[2] = (int)(4096u << 16);       // tensor_dim1 = 4096 (bits 111:80)
  g1[3] = (int)(64u << 16);         // tile_dim0 = 64 (bits 127:112)
  g1[4] = 32;                       // tile_dim1 = 32, tile_dim2 = 0
  g1[5] = 64;                       // tensor_dim0_stride = 64 (bits 207:160)
  g1[6] = 0;
  g1[7] = 0;
  v4i z4 = {0, 0, 0, 0};
#if __clang_major__ >= 23
  v8i z8 = {0, 0, 0, 0, 0, 0, 0, 0};
  __builtin_amdgcn_tensor_load_to_lds(g0, g1, z4, z4, z8, 0);
#else
  __builtin_amdgcn_tensor_load_to_lds(g0, g1, z4, z4, 0);
#endif
#endif
}

// ---------------------------------------------------------------------------
// Kernel 1: projections (memory-bound -> fp32 VALU math, bf16 outputs).
// Qt[b][n][c4] = (W_qk x)/sqrt(8) transposed; Vb[b][c4][n] = W_v x + b_v.
// ---------------------------------------------------------------------------
__global__ void __launch_bounds__(256)
proj_kernel(const float* __restrict__ x, const float* __restrict__ Wqk,
            const float* __restrict__ Wv, const float* __restrict__ bv,
            __bf16* __restrict__ Qt, __bf16* __restrict__ Vb) {
  const int n     = blockIdx.x * 64 + threadIdx.x;
  const int ot    = threadIdx.y;      // 16-output tile (wave-uniform)
  const int which = blockIdx.y;       // 0 = Q, 1 = V
  const int b     = blockIdx.z;
  const float* W  = which ? Wv : Wqk;

  float acc[16];
#pragma unroll
  for (int i = 0; i < 16; ++i) acc[i] = 0.f;

  const float* xp = x + ((size_t)b * CC) * NN + n;
  const float* wp = W + (size_t)(ot * 16) * CC;
  for (int c = 0; c < CC; ++c) {
    float xv = xp[(size_t)c * NN];
#pragma unroll
    for (int i = 0; i < 16; ++i) acc[i] = fmaf(wp[i * CC + c], xv, acc[i]);
  }

  if (which == 0) {
    __bf16* q = Qt + ((size_t)b * NN + n) * C4 + ot * 16;
#pragma unroll
    for (int i = 0; i < 16; ++i) q[i] = (__bf16)(acc[i] * QSCALE);
  } else {
#pragma unroll
    for (int i = 0; i < 16; ++i) {
      float v = acc[i] + bv[ot * 16 + i];
      Vb[((size_t)b * C4 + ot * 16 + i) * NN + n] = (__bf16)v;
    }
  }
}

// ---------------------------------------------------------------------------
// Kernel 2: per-row softmax statistics of E. Each wave owns TWO 16-row strips
// (A-frags loaded once, each B-frag reused for 2 WMMAs). Lane-local max and
// shifted exp-sum accumulate with no cross-lane traffic in the loop; a single
// 16-lane reduction runs at the end.
// ---------------------------------------------------------------------------
__global__ void __launch_bounds__(128)
stats_kernel(const __bf16* __restrict__ Qt, float* __restrict__ rmax,
             float* __restrict__ rinv) {
  const int b     = blockIdx.y;
  const int wave  = threadIdx.x >> 5;
  const int lane  = threadIdx.x & 31;
  const int half  = lane >> 4;
  const int l15   = lane & 15;
  const int nbase = blockIdx.x * 128 + wave * 32;   // two 16-row strips

  const __bf16* Qb = Qt + (size_t)b * NN * C4;
  v16bf a0[2], a1[2];
#pragma unroll
  for (int s = 0; s < 2; ++s) {
    const __bf16* arow = Qb + (size_t)(nbase + s * 16 + l15) * C4;
    a0[s] = load_a_frag(arow, 0, half);
    a1[s] = load_a_frag(arow, 1, half);
  }

  float lmax[2][8], lsum[2][8];
#pragma unroll
  for (int s = 0; s < 2; ++s)
#pragma unroll
    for (int j = 0; j < 8; ++j) { lmax[s][j] = -INFINITY; lsum[s][j] = 0.f; }

  for (int m0 = 0; m0 < NN; m0 += 16) {
    const __bf16* brow = Qb + (size_t)(m0 + l15) * C4;
    __builtin_prefetch(brow + 16 * C4, 0, 1);
    v16bf b0 = load_b_frag(brow, 0, half);
    v16bf b1 = load_b_frag(brow, 1, half);
#pragma unroll
    for (int s = 0; s < 2; ++s) {
      v8f d = {0.f, 0.f, 0.f, 0.f, 0.f, 0.f, 0.f, 0.f};
      d = wmma_bf16(a0[s], b0, d);
      d = wmma_bf16(a1[s], b1, d);
#pragma unroll
      for (int j = 0; j < 8; ++j) {
        float e = d[j];
        lmax[s][j] = fmaxf(lmax[s][j], e);
        lsum[s][j] += __expf(e - ESHIFT);   // TRANS op, co-executes with XDL
      }
    }
  }

#pragma unroll
  for (int s = 0; s < 2; ++s)
#pragma unroll
    for (int j = 0; j < 8; ++j) {
      float M = lmax[s][j], S = lsum[s][j];
#pragma unroll
      for (int d = 1; d < 16; d <<= 1) {
        M = fmaxf(M, __shfl_xor(M, d, 32));
        S += __shfl_xor(S, d, 32);
      }
      if (l15 == 0) {
        int r = nbase + s * 16 + j + 8 * half;
        rmax[(size_t)b * NN + r] = M;
        rinv[(size_t)b * NN + r] = __expf(M - ESHIFT) / S;
      }
    }
}

// ---------------------------------------------------------------------------
// Kernel 3: out[:, m] = V . P. Each wave owns TWO 16-wide m-tiles, all 64
// channels. The 32x64 bf16 Qt tile consumed per K-step is shared by all 4
// waves, so it is staged into LDS by the Tensor Data Mover (double-buffered,
// wave 0 posts the D#, S_WAIT_TENSORCNT + barrier hands it to consumers) --
// cutting per-WGP Qt VMEM traffic 4x. P is staged transposed in LDS (one
// ds_store_b128 per lane/subtile, two ds_load_b128 reads).
// ---------------------------------------------------------------------------
__global__ void __launch_bounds__(128)
attend_kernel(const __bf16* __restrict__ Qt, const __bf16* __restrict__ Vb,
              const float* __restrict__ rmax, const float* __restrict__ rinv,
              float* __restrict__ out) {
  __shared__ __bf16 Pt[4][2][16][40];   // P staging, pitch 40 bf16 (80 B)
  __shared__ __bf16 Qtile[2][32 * 64];  // double-buffered TDM destination, 2x4KB

  const int b     = blockIdx.y;
  const int wave  = threadIdx.x >> 5;
  const int lane  = threadIdx.x & 31;
  const int half  = lane >> 4;
  const int l15   = lane & 15;
  const int mbase = blockIdx.x * 128 + wave * 32;   // two 16-wide m tiles

  const __bf16* Qb  = Qt + (size_t)b * NN * C4;
  const __bf16* Vbb = Vb + (size_t)b * C4 * NN;
  const float*  rm  = rmax + (size_t)b * NN;
  const float*  ri  = rinv + (size_t)b * NN;

  // E-tile B operands (columns m), fixed for the whole loop
  v16bf bm0[2], bm1[2];
#pragma unroll
  for (int mt = 0; mt < 2; ++mt) {
    const __bf16* mrow = Qb + (size_t)(mbase + mt * 16 + l15) * C4;
    bm0[mt] = load_b_frag(mrow, 0, half);
    bm1[mt] = load_b_frag(mrow, 1, half);
  }

  v8f acc[2][4];
#pragma unroll
  for (int mt = 0; mt < 2; ++mt)
#pragma unroll
    for (int cb = 0; cb < 4; ++cb)
#pragma unroll
      for (int j = 0; j < 8; ++j) acc[mt][cb][j] = 0.f;

  const int niter = NN / 32;
#ifdef HAVE_TDM
  if (wave == 0) tdm_load_qtile(Qb, (unsigned)(uintptr_t)&Qtile[0][0]);
#else
  // fallback: wave 0 cooperative copy of tile 0
  if (wave == 0) {
#pragma unroll
    for (int k = 0; k < 8; ++k)
      *(v8bf*)&Qtile[0][(lane + k * 32) * 8] =
          *(const v8bf*)(Qb + (size_t)(lane + k * 32) * 8);
  }
#endif

  for (int i = 0; i < niter; ++i) {
    const int buf = i & 1;
    if (wave == 0) {
      if (i + 1 < niter) {
#ifdef HAVE_TDM
        tdm_load_qtile(Qb + (size_t)(i + 1) * 32 * C4,
                       (unsigned)(uintptr_t)&Qtile[buf ^ 1][0]);
        __builtin_amdgcn_s_wait_tensorcnt(1);  // tile i done, i+1 in flight
#else
#pragma unroll
        for (int k = 0; k < 8; ++k)
          *(v8bf*)&Qtile[buf ^ 1][(lane + k * 32) * 8] =
              *(const v8bf*)(Qb + (size_t)((i + 1) * 32 + lane + k * 32) * 8);
#endif
      } else {
#ifdef HAVE_TDM
        __builtin_amdgcn_s_wait_tensorcnt(0);
#endif
      }
    }
    __syncthreads();   // tile i visible to all waves

    const __bf16* qt = &Qtile[buf][0];
    const int n0 = i * 32;
#pragma unroll
    for (int s = 0; s < 2; ++s) {
      const int nb = n0 + s * 16;
      // A frags from the LDS-staged tile (ds_load_b128 pairs)
      const __bf16* arow = qt + (size_t)(s * 16 + l15) * C4;
      v16bf a0 = load_a_frag(arow, 0, half);
      v16bf a1 = load_a_frag(arow, 1, half);
      float rmv[8], riv[8];
#pragma unroll
      for (int j = 0; j < 8; ++j) {
        int r = nb + j + 8 * half;
        rmv[j] = rm[r];
        riv[j] = ri[r];
      }
#pragma unroll
      for (int mt = 0; mt < 2; ++mt) {
        v8f d = {0.f, 0.f, 0.f, 0.f, 0.f, 0.f, 0.f, 0.f};
        d = wmma_bf16(a0, bm0[mt], d);
        d = wmma_bf16(a1, bm1[mt], d);
        v8bf pk;
#pragma unroll
        for (int j = 0; j < 8; ++j)
          pk[j] = (__bf16)(__expf(d[j] - rmv[j]) * riv[j]);
        // lane's 8 elements: column m = l15, rows nb+8*half+0..7 -> one 16B store
        *(v8bf*)&Pt[wave][mt][l15][s * 16 + 8 * half] = pk;
      }
    }
    asm volatile("s_wait_dscnt 0" ::: "memory");  // within-wave P turnaround

    V16U pb[2];
#pragma unroll
    for (int mt = 0; mt < 2; ++mt) {
      const v8bf* pp = (const v8bf*)&Pt[wave][mt][l15][half * 16];
      pb[mt].h[0] = pp[0];
      pb[mt].h[1] = pp[1];
    }

#pragma unroll
    for (int cb = 0; cb < 4; ++cb) {
      const __bf16* vrow = Vbb + (size_t)(cb * 16 + l15) * NN;
      V16U av;   // V fragment shared by both m-tiles
      av.h[0] = *(const v8bf*)(vrow + n0 + half * 16);
      av.h[1] = *(const v8bf*)(vrow + n0 + half * 16 + 8);
#pragma unroll
      for (int mt = 0; mt < 2; ++mt)
        acc[mt][cb] = wmma_bf16(av.v, pb[mt].v, acc[mt][cb]);
    }
    __syncthreads();   // all reads of buf done before it is refilled
  }

  float* ob = out + (size_t)b * C4 * NN;
#pragma unroll
  for (int mt = 0; mt < 2; ++mt)
#pragma unroll
    for (int cb = 0; cb < 4; ++cb)
#pragma unroll
      for (int j = 0; j < 8; ++j)
        ob[(size_t)(cb * 16 + j + 8 * half) * NN + mbase + mt * 16 + l15] =
            acc[mt][cb][j];
}

// ---------------------------------------------------------------------------
extern "C" void kernel_launch(void* const* d_in, const int* in_sizes, int n_in,
                              void* d_out, int out_size, void* d_ws, size_t ws_size,
                              hipStream_t stream) {
  const float* x   = (const float*)d_in[0]; // [B, C, N]
  const float* Wqk = (const float*)d_in[1]; // [C4, C]
  const float* Wv  = (const float*)d_in[2]; // [C4, C]
  const float* bv  = (const float*)d_in[3]; // [C4]
  float* out = (float*)d_out;               // [B, C4, N]

  // Workspace: Qt (2 MB bf16) | Vb (2 MB bf16) | rmax (64 KB) | rinv (64 KB)
  __bf16* Qt  = (__bf16*)d_ws;
  __bf16* Vb  = Qt + (size_t)BB * NN * C4;
  float* rmax = (float*)(Vb + (size_t)BB * NN * C4);
  float* rinv = rmax + (size_t)BB * NN;

  dim3 g1(NN / 64, 2, BB), b1(64, 4, 1);
  proj_kernel<<<g1, b1, 0, stream>>>(x, Wqk, Wv, bv, Qt, Vb);

  dim3 g2(NN / 128, BB);
  stats_kernel<<<g2, 128, 0, stream>>>(Qt, rmax, rinv);

  dim3 g3(NN / 128, BB);
  attend_kernel<<<g3, 128, 0, stream>>>(Qt, Vb, rmax, rinv, out);
}
